// feature_axial_MultiHeadSelfAttention_49941879718277
// MI455X (gfx1250) — matrix operand order repaired and checked by reading, not compiled, not run on hardware
//
#include <hip/hip_runtime.h>

// ---------------------------------------------------------------------------
// Feature-axial multi-head self-attention for MI455X (gfx1250, wave32, WMMA).
// Heavy math via v_wmma_f32_16x16x32_bf16; global->LDS via async-LDS loads
// (ASYNCcnt) with double buffering. 64x64 wave tiles: 16 WMMA per K-step per
// wave against 16 ds_load_b128 (best LDS-traffic/FLOP at this block size).
// ---------------------------------------------------------------------------

typedef __attribute__((ext_vector_type(16))) __bf16 v16bf;
typedef __attribute__((ext_vector_type(8)))  float  v8f;

#define BLK_M 128
#define BLK_N 128
#define NTHREADS 128     // 4 waves, arranged 2(M) x 2(N), 64x64 tile per wave
#define LDS_STRIDE 40    // halfwords per LDS row (32 data + 8 pad -> 80B, 16B-aligned rows)
#define LDS_STRIDE_U 20  // same, in dword units
#define TILE_HW (BLK_M * LDS_STRIDE)

#if __has_builtin(__builtin_amdgcn_global_load_async_to_lds_b128)
#define HAVE_ASYNC_LDS 1
// Builtin signature (probe-confirmed): (int4 AS1*, int4 AS3*, imm int, imm int)
typedef int i4_t __attribute__((ext_vector_type(4)));
typedef __attribute__((address_space(1))) i4_t* GP128;
typedef __attribute__((address_space(3))) i4_t* LP128;
#else
#define HAVE_ASYNC_LDS 0
#endif

#if __has_builtin(__builtin_amdgcn_s_wait_asynccnt)
#define WAIT_ASYNCCNT(n) __builtin_amdgcn_s_wait_asynccnt(n)
#else
#define WAIT_ASYNCCNT(n) asm volatile("s_wait_asynccnt %0" ::"i"(n) : "memory")
#endif

union Frag {
    unsigned long long d[4];
    v16bf v;
};

__device__ __forceinline__ unsigned short f32_to_bf16(float f) {
    unsigned u = __float_as_uint(f);
    u += 0x7FFFu + ((u >> 16) & 1u);   // round to nearest even
    return (unsigned short)(u >> 16);
}

// ---------------------------------------------------------------------------
// fp32 -> bf16 elementwise conversion (4 elems / thread)
// ---------------------------------------------------------------------------
__global__ __launch_bounds__(256) void cvt_f32_bf16(const float* __restrict__ in,
                                                    unsigned short* __restrict__ out,
                                                    int n) {
    int i = (blockIdx.x * 256 + threadIdx.x) * 4;
    if (i >= n) return;
    float4 f = *(const float4*)(in + i);
    unsigned r0 = (unsigned)f32_to_bf16(f.x) | ((unsigned)f32_to_bf16(f.y) << 16);
    unsigned r1 = (unsigned)f32_to_bf16(f.z) | ((unsigned)f32_to_bf16(f.w) << 16);
    *(uint2*)(out + i) = make_uint2(r0, r1);
}

// ---------------------------------------------------------------------------
// Tiled WMMA GEMM, NT form:  C[m,n] = sum_k A[m,k] * B[n,k]
//   MODE 0: C f32  row-major [M x N]   MODE 1: C bf16 row-major [M x N]
//   MODE 2: C bf16 TRANSPOSED [N x M] (Ct[n*ldc + m]) via LDS transpose
// 128 threads (4 waves, 2x2), block tile 128x128, K-step 32.
// Wave tile 64x64 = 16 WMMA per K-step (4 A-frags x 4 B-frags).
// M,N multiples of 128; Kd multiple of 32 (true for all calls here).
// ---------------------------------------------------------------------------
template <int MODE>
__global__ __launch_bounds__(NTHREADS)
void gemm_bf16_nt(const unsigned short* __restrict__ A, int lda,
                  const unsigned short* __restrict__ B, int ldb,
                  void* __restrict__ Cout, int ldc, int Kd) {
#if HAVE_ASYNC_LDS
    __shared__ unsigned short As[2 * TILE_HW];   // double buffered
    __shared__ unsigned short Bs[2 * TILE_HW];
#else
    __shared__ unsigned short As[TILE_HW];
    __shared__ unsigned short Bs[TILE_HW];
#endif
    __shared__ unsigned short Ct[(MODE == 2) ? (BLK_N * (BLK_M + 8)) : 2];

    const int tid  = threadIdx.x;
    const int lane = tid & 31;
    const int wave = tid >> 5;
    const int wm = wave & 1;   // wave position along M (2)
    const int wn = wave >> 1;  // wave position along N (2)
    const int h  = lane >> 4;  // lane half (CDNA5 WMMA layout)
    const int ln = lane & 15;

    const int m0 = blockIdx.y * BLK_M;
    const int n0 = blockIdx.x * BLK_N;

    // Tile staging: 128 rows x 32 halfwords = 512 x 16B chunks per operand;
    // each thread owns 4 chunks of A and 4 chunks of B.
    const unsigned short* Agp[4];
    const unsigned short* Bgp[4];
    int shh[4];
    #pragma unroll
    for (int i = 0; i < 4; ++i) {
        const int c = tid + i * NTHREADS;
        const int r = c >> 2, cc = (c & 3) << 3;
        Agp[i] = A + (size_t)(m0 + r) * lda + cc;
        Bgp[i] = B + (size_t)(n0 + r) * ldb + cc;
        shh[i] = r * LDS_STRIDE + cc;  // halfword offset inside a tile buffer
    }

    const int nk = Kd >> 5;
    v8f acc[4][4] = {};

#if HAVE_ASYNC_LDS
    // ---- async double-buffered pipeline (gfx1250 GLOBAL_LOAD_ASYNC_TO_LDS) --
    #pragma unroll
    for (int i = 0; i < 4; ++i) {
        __builtin_amdgcn_global_load_async_to_lds_b128((GP128)Agp[i], (LP128)(As + shh[i]), 0, 0);
        __builtin_amdgcn_global_load_async_to_lds_b128((GP128)Bgp[i], (LP128)(Bs + shh[i]), 0, 0);
    }

    for (int kt = 0; kt < nk; ++kt) {
        const int cur = kt & 1;
        __syncthreads();  // all waves done reading buf[cur^1] (iteration kt-1)
        if (kt + 1 < nk) {
            const int ko  = (kt + 1) << 5;
            const int nxt = (cur ^ 1) * TILE_HW;
            #pragma unroll
            for (int i = 0; i < 4; ++i) {
                __builtin_amdgcn_global_load_async_to_lds_b128((GP128)(Agp[i] + ko), (LP128)(As + nxt + shh[i]), 0, 0);
                __builtin_amdgcn_global_load_async_to_lds_b128((GP128)(Bgp[i] + ko), (LP128)(Bs + nxt + shh[i]), 0, 0);
            }
            WAIT_ASYNCCNT(8);  // 8 newer outstanding => tile kt's 8 loads landed
        } else {
            WAIT_ASYNCCNT(0);
        }
        __syncthreads();  // everyone's tile-kt chunks are in LDS

        const unsigned* AsU = (const unsigned*)(As + cur * TILE_HW);
        const unsigned* BsU = (const unsigned*)(Bs + cur * TILE_HW);
#else
    // ---- register-staged fallback pipeline ---------------------------------
    unsigned* AsU = (unsigned*)As;
    unsigned* BsU = (unsigned*)Bs;
    uint4 ra[4], rb[4];
    #pragma unroll
    for (int i = 0; i < 4; ++i) {
        ra[i] = *(const uint4*)Agp[i];
        rb[i] = *(const uint4*)Bgp[i];
    }

    for (int kt = 0; kt < nk; ++kt) {
        __syncthreads();
        #pragma unroll
        for (int i = 0; i < 4; ++i) {
            const int sa = (shh[i] >> 1);
            *(uint2*)(AsU + sa)     = make_uint2(ra[i].x, ra[i].y);
            *(uint2*)(AsU + sa + 2) = make_uint2(ra[i].z, ra[i].w);
            *(uint2*)(BsU + sa)     = make_uint2(rb[i].x, rb[i].y);
            *(uint2*)(BsU + sa + 2) = make_uint2(rb[i].z, rb[i].w);
        }
        __syncthreads();
        if (kt + 1 < nk) {
            const int ko = (kt + 1) << 5;
            #pragma unroll
            for (int i = 0; i < 4; ++i) {
                ra[i] = *(const uint4*)(Agp[i] + ko);
                rb[i] = *(const uint4*)(Bgp[i] + ko);
            }
        }
#endif
        // Assemble WMMA fragments from LDS per CDNA5 layouts.
        // A 16x32 bf16: VGPR v<4 -> K = 2v + 8h ; v>=4 -> K = 16 + 2(v-4) + 8h
        //   => dword idx within row: v + 4h (v<4), 8 + (v-4) + 4h (v>=4)
        Frag af[4], bf[4];
        #pragma unroll
        for (int ms = 0; ms < 4; ++ms) {
            const int base = (wm * 64 + ms * 16 + ln) * LDS_STRIDE_U + (h << 2);
            af[ms].d[0] = *(const unsigned long long*)(AsU + base);
            af[ms].d[1] = *(const unsigned long long*)(AsU + base + 2);
            af[ms].d[2] = *(const unsigned long long*)(AsU + base + 8);
            af[ms].d[3] = *(const unsigned long long*)(AsU + base + 10);
        }
        // B 32x16 bf16: same K order as A
        //   => dword idx within row: v + 4h (v<4), 8 + (v-4) + 4h (v>=4)
        #pragma unroll
        for (int ns = 0; ns < 4; ++ns) {
            const int base = (wn * 64 + ns * 16 + ln) * LDS_STRIDE_U + (h << 2);
            bf[ns].d[0] = *(const unsigned long long*)(BsU + base);
            bf[ns].d[1] = *(const unsigned long long*)(BsU + base + 2);
            bf[ns].d[2] = *(const unsigned long long*)(BsU + base + 8);
            bf[ns].d[3] = *(const unsigned long long*)(BsU + base + 10);
        }

        #pragma unroll
        for (int ms = 0; ms < 4; ++ms)
            #pragma unroll
            for (int ns = 0; ns < 4; ++ns)
                acc[ms][ns] = __builtin_amdgcn_wmma_f32_16x16x32_bf16(
                    false, af[ms].v, false, bf[ns].v,
                    (short)0, acc[ms][ns], false, false);
    }

    // Epilogue. C layout: VGPR v holds C[M = v + 8h][N = ln].
    if (MODE == 0) {
        float* C = (float*)Cout;
        #pragma unroll
        for (int ms = 0; ms < 4; ++ms)
            #pragma unroll
            for (int ns = 0; ns < 4; ++ns) {
                const int mg = m0 + wm * 64 + ms * 16 + h * 8;
                const int ng = n0 + wn * 64 + ns * 16 + ln;
                #pragma unroll
                for (int v = 0; v < 8; ++v)
                    C[(size_t)(mg + v) * ldc + ng] = acc[ms][ns][v];
            }
    } else if (MODE == 1) {
        unsigned short* C = (unsigned short*)Cout;
        #pragma unroll
        for (int ms = 0; ms < 4; ++ms)
            #pragma unroll
            for (int ns = 0; ns < 4; ++ns) {
                const int mg = m0 + wm * 64 + ms * 16 + h * 8;
                const int ng = n0 + wn * 64 + ns * 16 + ln;
                #pragma unroll
                for (int v = 0; v < 8; ++v)
                    C[(size_t)(mg + v) * ldc + ng] = f32_to_bf16(acc[ms][ns][v]);
            }
    } else {
        // Transposed bf16 store via LDS so global writes stay coalesced.
        #pragma unroll
        for (int ms = 0; ms < 4; ++ms)
            #pragma unroll
            for (int ns = 0; ns < 4; ++ns) {
                const int ml = wm * 64 + ms * 16 + h * 8;
                const int nl = wn * 64 + ns * 16 + ln;
                #pragma unroll
                for (int v = 0; v < 8; ++v)
                    Ct[nl * (BLK_M + 8) + ml + v] = f32_to_bf16(acc[ms][ns][v]);
            }
        __syncthreads();
        unsigned short* C = (unsigned short*)Cout;  // Ct[n * ldc + m], ldc = total M
        for (int i = tid; i < BLK_N * 16; i += NTHREADS) {
            const int r = i >> 4, ch = i & 15;
            uint4 vdat = *(const uint4*)(Ct + r * (BLK_M + 8) + ch * 8);
            *(uint4*)(C + (size_t)(n0 + r) * ldc + m0 + ch * 8) = vdat;
        }
    }
}

// ---------------------------------------------------------------------------
// Row softmax: in f32 rows of length 1024, out bf16. One 256-thread block/row.
// Applies scale (1/sqrt(K)) before max-subtraction.
// ---------------------------------------------------------------------------
__global__ __launch_bounds__(256)
void softmax_rows(const float* __restrict__ in, unsigned short* __restrict__ out,
                  int n, float scale) {
    __shared__ float red[256];
    const int tid = threadIdx.x;
    const float* row = in + (size_t)blockIdx.x * n;
    unsigned short* orow = out + (size_t)blockIdx.x * n;

    float4 x = *(const float4*)(row + tid * 4);
    float a0 = x.x * scale, a1 = x.y * scale, a2 = x.z * scale, a3 = x.w * scale;

    float m = fmaxf(fmaxf(a0, a1), fmaxf(a2, a3));
    red[tid] = m;
    __syncthreads();
    for (int s = 128; s > 0; s >>= 1) {
        if (tid < s) red[tid] = fmaxf(red[tid], red[tid + s]);
        __syncthreads();
    }
    const float rm = red[0];
    __syncthreads();

    float e0 = __expf(a0 - rm), e1 = __expf(a1 - rm);
    float e2 = __expf(a2 - rm), e3 = __expf(a3 - rm);
    red[tid] = e0 + e1 + e2 + e3;
    __syncthreads();
    for (int s = 128; s > 0; s >>= 1) {
        if (tid < s) red[tid] += red[tid + s];
        __syncthreads();
    }
    const float inv = 1.0f / red[0];

    unsigned r0 = (unsigned)f32_to_bf16(e0 * inv) | ((unsigned)f32_to_bf16(e1 * inv) << 16);
    unsigned r1 = (unsigned)f32_to_bf16(e2 * inv) | ((unsigned)f32_to_bf16(e3 * inv) << 16);
    *(uint2*)(orow + tid * 4) = make_uint2(r0, r1);
}

// ---------------------------------------------------------------------------
// Host-side orchestration (stream-ordered only; workspace carved from d_ws)
// ---------------------------------------------------------------------------
extern "C" void kernel_launch(void* const* d_in, const int* in_sizes, int n_in,
                              void* d_out, int out_size, void* d_ws, size_t ws_size,
                              hipStream_t stream) {
    (void)in_sizes; (void)n_in; (void)out_size; (void)ws_size;

    const float* x  = (const float*)d_in[0];
    const float* Wq = (const float*)d_in[1];
    const float* Wk = (const float*)d_in[2];
    const float* Wv = (const float*)d_in[3];
    const float* Wo = (const float*)d_in[4];

    const int B = 4, L = 4096, E = 1024, K = 1024;
    const int ML = B * L;  // 16384

    size_t off = 0;
    char* ws = (char*)d_ws;
    auto alloc = [&](size_t bytes) -> void* {
        void* p = ws + off;
        off += (bytes + 255) & ~(size_t)255;
        return p;
    };
    unsigned short* xbf  = (unsigned short*)alloc((size_t)ML * E * 2);
    unsigned short* wqb  = (unsigned short*)alloc((size_t)K * E * 2);
    unsigned short* wkb  = (unsigned short*)alloc((size_t)K * E * 2);
    unsigned short* wvb  = (unsigned short*)alloc((size_t)K * E * 2);
    unsigned short* wob  = (unsigned short*)alloc((size_t)E * K * 2);
    unsigned short* qT   = (unsigned short*)alloc((size_t)K * ML * 2);  // [K][B*L]
    unsigned short* kT   = (unsigned short*)alloc((size_t)K * ML * 2);  // [K][B*L]
    unsigned short* vbf  = (unsigned short*)alloc((size_t)ML * K * 2);  // [B*L][K]
    float*          attf = (float*)alloc((size_t)B * K * K * 4);
    unsigned short* attb = (unsigned short*)alloc((size_t)B * K * K * 2);
    unsigned short* outb = (unsigned short*)alloc((size_t)ML * K * 2);

    // 1) fp32 -> bf16 conversions
    cvt_f32_bf16<<<(ML * E / 4 + 255) / 256, 256, 0, stream>>>(x, xbf, ML * E);
    cvt_f32_bf16<<<(K * E / 4 + 255) / 256, 256, 0, stream>>>(Wq, wqb, K * E);
    cvt_f32_bf16<<<(K * E / 4 + 255) / 256, 256, 0, stream>>>(Wk, wkb, K * E);
    cvt_f32_bf16<<<(K * E / 4 + 255) / 256, 256, 0, stream>>>(Wv, wvb, K * E);
    cvt_f32_bf16<<<(E * K / 4 + 255) / 256, 256, 0, stream>>>(Wo, wob, E * K);

    // 2) q/k projections with transposed bf16 store: qT[kf][b*L+l]
    gemm_bf16_nt<2><<<dim3(K / BLK_N, ML / BLK_M), NTHREADS, 0, stream>>>(
        xbf, E, wqb, E, qT, ML, E);
    gemm_bf16_nt<2><<<dim3(K / BLK_N, ML / BLK_M), NTHREADS, 0, stream>>>(
        xbf, E, wkb, E, kT, ML, E);
    // 3) v projection, bf16 row-major
    gemm_bf16_nt<1><<<dim3(K / BLK_N, ML / BLK_M), NTHREADS, 0, stream>>>(
        xbf, E, wvb, E, vbf, K, E);

    // 4) att[b][kf][j] = sum_l qT[kf][bL+l] * kT[j][bL+l]  (f32 out)
    for (int b = 0; b < B; ++b)
        gemm_bf16_nt<0><<<dim3(K / BLK_N, K / BLK_M), NTHREADS, 0, stream>>>(
            qT + (size_t)b * L, ML, kT + (size_t)b * L, ML,
            attf + (size_t)b * K * K, K, L);

    // 5) softmax over j with scale 1/sqrt(K) = 1/32, bf16 out
    softmax_rows<<<B * K, 256, 0, stream>>>(attf, attb, K, 1.0f / 32.0f);

    // 6) out[b][l][kf] = sum_j v[b][l][j] * att[b][kf][j]  (bf16 out)
    for (int b = 0; b < B; ++b)
        gemm_bf16_nt<1><<<dim3(K / BLK_N, L / BLK_M), NTHREADS, 0, stream>>>(
            vbf + (size_t)b * L * K, K, attb + (size_t)b * K * K, K,
            outb + (size_t)b * L * K, K, K);

    // 7) y[b][l][e] = sum_v out[b][l][v] * Wo[e][v]  (f32 out -> d_out)
    gemm_bf16_nt<0><<<dim3(E / BLK_N, ML / BLK_M), NTHREADS, 0, stream>>>(
        outb, K, wob, K, (float*)d_out, E, K);
}
